// AttentionBlock_3994319586008
// MI455X (gfx1250) — compile-verified, matrix operand order checked by reading, not run on hardware
//
#include <hip/hip_runtime.h>
#include <stdint.h>

// ---------------------------------------------------------------------------
// CDNA5 / gfx1250 attention block: GroupNorm -> QKV (WMMA) -> flash attention
// (WMMA, async-LDS staged K/V) -> proj (WMMA) + bias + residual.
// wave32, bf16 operands, f32 accumulate.
// ---------------------------------------------------------------------------

typedef __attribute__((ext_vector_type(16))) __bf16    v16bf;
typedef __attribute__((ext_vector_type(8)))  float     v8f;
typedef __attribute__((ext_vector_type(4)))  uint32_t  u32x4;
typedef __attribute__((ext_vector_type(4)))  float     f32x4;

#define BATCH 32
#define CCH   128
#define HWN   1024
#define NH    4
#define DH    32
#define QKSCALE 0.17677669529663687f  // 1/sqrt(32)

union FragU { u32x4 q2[2]; v16bf v; };

__device__ __forceinline__ uint16_t f2bf(float f) {
    uint32_t u = __builtin_bit_cast(uint32_t, f);
    u += 0x7fffu + ((u >> 16) & 1u);          // round-to-nearest-even
    return (uint16_t)(u >> 16);
}

// Fragment load: per-lane 8 bf16 at base and 8 bf16 at base+16 elements.
// Caller passes base already offset by half*8 so lane halves get the ISA's
// K split {0..7,16..23} / {8..15,24..31}.
__device__ __forceinline__ v16bf load_frag(const uint16_t* p) {
    FragU f;
    f.q2[0] = *(const u32x4*)(p);
    f.q2[1] = *(const u32x4*)(p + 16);
    return f.v;
}

__device__ __forceinline__ v8f wmma_bf16(v16bf a, v16bf b, v8f c) {
    return __builtin_amdgcn_wmma_f32_16x16x32_bf16(
        false, a, false, b, (short)0, c, false, false);
}

// Pack 8 accumulator floats (scaled) to 8 bf16 and store as one b128.
__device__ __forceinline__ void store8bf(uint16_t* dst, v8f a, float s) {
    uint32_t w[4];
#pragma unroll
    for (int i = 0; i < 4; ++i)
        w[i] = (uint32_t)f2bf(a[2 * i] * s) | ((uint32_t)f2bf(a[2 * i + 1] * s) << 16);
    u32x4 pk = {w[0], w[1], w[2], w[3]};
    *(u32x4*)dst = pk;
}

// Row reductions across a 16-lane half (xor masks <= 8 stay inside halves).
__device__ __forceinline__ float hmax16(float v) {
    v = fmaxf(v, __shfl_xor(v, 8, 32));
    v = fmaxf(v, __shfl_xor(v, 4, 32));
    v = fmaxf(v, __shfl_xor(v, 2, 32));
    v = fmaxf(v, __shfl_xor(v, 1, 32));
    return v;
}
__device__ __forceinline__ float hsum16(float v) {
    v += __shfl_xor(v, 8, 32);
    v += __shfl_xor(v, 4, 32);
    v += __shfl_xor(v, 2, 32);
    v += __shfl_xor(v, 1, 32);
    return v;
}

// Async DMA: one 16-byte global->LDS transfer per active lane (ASYNCcnt).
__device__ __forceinline__ void async_b128(uint16_t* lds_dst, const uint16_t* gsrc) {
    uint32_t la = (uint32_t)(uintptr_t)lds_dst;          // low 32 bits = LDS offset
    uint64_t ga = (uint64_t)(uintptr_t)gsrc;
    asm volatile("global_load_async_to_lds_b128 %0, %1, off"
                 :: "v"(la), "v"(ga) : "memory");
}

// ------------------------- kernel 0: weight convert -------------------------
__global__ void k_wconvert(const float* __restrict__ qkv_w,
                           const float* __restrict__ proj_w,
                           uint16_t* __restrict__ qkv_bf,
                           uint16_t* __restrict__ proj_bf) {
    int i = blockIdx.x * 256 + threadIdx.x;
    if (i < 3 * CCH * CCH) qkv_bf[i] = f2bf(qkv_w[i]);
    if (i < CCH * CCH)     proj_bf[i] = f2bf(proj_w[i]);
}

// ------------------------- kernel 1: GroupNorm ------------------------------
// One 256-thread block per (b, group). Group = 4 channels x 1024 spatial.
// Writes xn transposed to (b, t, c) bf16 (GEMM-B friendly layout).
__global__ void k_groupnorm(const float* __restrict__ x,
                            const float* __restrict__ nw,
                            const float* __restrict__ nb,
                            uint16_t* __restrict__ xn) {
    __shared__ float ssum[256];
    __shared__ float ssq[256];
    int tid = threadIdx.x;
    int b = blockIdx.x >> 5;
    int g = blockIdx.x & 31;
    int cl = tid >> 6;               // 0..3 channel within group
    int t0 = (tid & 63) * 16;        // 16 consecutive spatial positions
    int c  = g * 4 + cl;

    const float* xp = x + ((size_t)(b * CCH + c)) * HWN + t0;
    float vals[16];
    float s = 0.f, sq = 0.f;
#pragma unroll
    for (int j = 0; j < 4; ++j) {
        f32x4 v4 = *(const f32x4*)(xp + j * 4);
#pragma unroll
        for (int e = 0; e < 4; ++e) {
            float f = v4[e];
            vals[j * 4 + e] = f;
            s += f;
            sq += f * f;
        }
    }
    ssum[tid] = s; ssq[tid] = sq;
    __syncthreads();
    for (int off = 128; off > 0; off >>= 1) {
        if (tid < off) { ssum[tid] += ssum[tid + off]; ssq[tid] += ssq[tid + off]; }
        __syncthreads();
    }
    float mu   = ssum[0] * (1.f / 4096.f);
    float var  = ssq[0] * (1.f / 4096.f) - mu * mu;
    float rstd = rsqrtf(var + 1e-5f);
    float wch  = nw[c] * rstd;
    float bch  = nb[c] - mu * wch;

    uint16_t* op = xn + ((size_t)b * HWN + t0) * CCH + c;
#pragma unroll
    for (int j = 0; j < 16; ++j)
        op[j * CCH] = f2bf(vals[j] * wch + bch);
}

// ------------------------- kernel 2: QKV GEMM -------------------------------
// Per batch: (384 x 128) @ (128 x 1024).  One wave computes a 16x64 strip.
// Writes q,k as (b,h,t,dh) scaled by 1/sqrt(dh); v as (b,h,dh,t).
// Tile routing (q/k/v section, head) is wave-uniform -> scalar branches.
__global__ void k_qkv_gemm(const uint16_t* __restrict__ wbf,
                           const uint16_t* __restrict__ xn,
                           uint16_t* __restrict__ q,
                           uint16_t* __restrict__ k,
                           uint16_t* __restrict__ v) {
    int wave = __builtin_amdgcn_readfirstlane((int)blockIdx.x * 8 + (threadIdx.x >> 5));
    int lane = threadIdx.x & 31;
    int half = lane >> 4, lm = lane & 15;
    int b   = wave / 384;
    int rem = wave - b * 384;
    int mt  = rem >> 4;               // 0..23  (16-row tile of 384)
    int n0  = (rem & 15) * 64;        // column strip start

    int head = mt / 6;                // all wave-uniform scalars
    int mt6  = mt - head * 6;
    int sec  = mt6 >> 1;              // 0=q 1=k 2=v
    int c0   = (mt6 & 1) * 16;
    size_t bh = (size_t)b * NH + head;

    v16bf A[4];
    const uint16_t* arow = wbf + (size_t)(mt * 16 + lm) * CCH + half * 8;
#pragma unroll
    for (int kk = 0; kk < 4; ++kk) A[kk] = load_frag(arow + kk * 32);

#pragma unroll
    for (int nt = 0; nt < 4; ++nt) {
        int t = n0 + nt * 16 + lm;
        const uint16_t* brow = xn + ((size_t)b * HWN + t) * CCH + half * 8;
        if (nt < 3) __builtin_prefetch(brow + 16 * CCH, 0, 3);
        v8f acc = {};
#pragma unroll
        for (int kk = 0; kk < 4; ++kk)
            acc = wmma_bf16(A[kk], load_frag(brow + kk * 32), acc);

        if (sec == 2) {
            uint16_t* dst = v + (bh * DH + c0 + half * 8) * HWN + t;
#pragma unroll
            for (int r = 0; r < 8; ++r)
                dst[(size_t)r * HWN] = f2bf(acc[r]);
        } else {
            uint16_t* dst = (sec == 0 ? q : k) + (bh * HWN + t) * DH + c0 + half * 8;
            store8bf(dst, acc, QKSCALE);       // one b128 store per lane
        }
    }
}

// ------------------------- kernel 3: flash attention ------------------------
// One block = one (b, h, group of 8 token tiles); 8 waves, one 16-row tile
// each.  Wave 0 DMA-stages shared K/V 32-key chunks into LDS with
// global_load_async_to_lds_b128 (double buffered, ASYNCcnt).  Online softmax
// with per-row stats in lane halves; P re-laid-out D-frag -> A-frag via LDS.
__global__ void k_attention(const uint16_t* __restrict__ q,
                            const uint16_t* __restrict__ k,
                            const uint16_t* __restrict__ v,
                            uint16_t* __restrict__ ho) {
    __shared__ __align__(16) uint16_t sK[2][32 * 32];   // (s_local, c)
    __shared__ __align__(16) uint16_t sV[2][32 * 32];   // (c, s_local)
    __shared__ __align__(16) uint16_t sP[8][16 * 32];

    int tid  = threadIdx.x;
    int wib  = __builtin_amdgcn_readfirstlane(tid >> 5);
    int lane = tid & 31;
    int half = lane >> 4, lm = lane & 15;
    int row  = lane >> 2, seg = lane & 3;     // DMA lane mapping: 8 rows x 4 segs

    int blk = blockIdx.x;                     // 32*4*8 blocks
    int b   = blk >> 5;
    int rem = blk & 31;
    int h   = rem >> 3;
    int tg  = rem & 7;
    int t0  = (tg * 8 + wib) * 16;
    size_t bh = (size_t)b * NH + h;
    const uint16_t* kbase = k + bh * HWN * DH;
    const uint16_t* vbase = v + bh * DH * HWN;

    v16bf Qf = load_frag(q + (bh * HWN + t0 + lm) * DH + half * 8);
    v8f O0 = {}, O1 = {};
    float m[8], l[8];
#pragma unroll
    for (int r = 0; r < 8; ++r) { m[r] = -1e30f; l[r] = 0.f; }

    uint16_t* lp = sP[wib];

    // Prologue: stage chunk 0.
    if (wib == 0) {
#pragma unroll
        for (int i = 0; i < 4; ++i) {
            int rr = i * 8 + row;
            async_b128(&sK[0][rr * 32 + seg * 8], kbase + (size_t)rr * DH + seg * 8);
            async_b128(&sV[0][rr * 32 + seg * 8], vbase + (size_t)rr * HWN + seg * 8);
        }
        asm volatile("s_wait_asynccnt 0x0" ::: "memory");
    }
    __syncthreads();

    for (int it = 0; it < HWN / 32; ++it) {
        int cur = it & 1;
        if (wib == 0 && it + 1 < HWN / 32) {
            int sn = (it + 1) * 32;
#pragma unroll
            for (int i = 0; i < 4; ++i) {
                int rr = i * 8 + row;
                async_b128(&sK[cur ^ 1][rr * 32 + seg * 8],
                           kbase + (size_t)(sn + rr) * DH + seg * 8);
                async_b128(&sV[cur ^ 1][rr * 32 + seg * 8],
                           vbase + (size_t)rr * HWN + sn + seg * 8);
            }
        }

        const uint16_t* kc = sK[cur];
        const uint16_t* vc = sV[cur];
        v16bf K0 = load_frag(kc + lm * 32 + half * 8);          // s = 0..15
        v16bf K1 = load_frag(kc + (16 + lm) * 32 + half * 8);   // s = 16..31
        v8f S0 = {}, S1 = {};
        S0 = wmma_bf16(Qf, K0, S0);
        S1 = wmma_bf16(Qf, K1, S1);

#pragma unroll
        for (int r = 0; r < 8; ++r) {
            float rowm = hmax16(fmaxf(S0[r], S1[r]));
            float newm = fmaxf(m[r], rowm);
            float f    = __expf(m[r] - newm);
            float p0   = __expf(S0[r] - newm);
            float p1   = __expf(S1[r] - newm);
            l[r] = l[r] * f + hsum16(p0 + p1);
            m[r] = newm;
            O0[r] *= f;
            O1[r] *= f;
            lp[(r + half * 8) * 32 + lm]      = f2bf(p0);
            lp[(r + half * 8) * 32 + 16 + lm] = f2bf(p1);
        }
        asm volatile("s_wait_dscnt 0x0" ::: "memory");   // LDS turnaround

        v16bf Pf = load_frag(lp + lm * 32 + half * 8);   // A-frag 16(t) x 32(s)
        v16bf V0 = load_frag(vc + lm * 32 + half * 8);          // c = 0..15
        v16bf V1 = load_frag(vc + (16 + lm) * 32 + half * 8);   // c = 16..31
        O0 = wmma_bf16(Pf, V0, O0);
        O1 = wmma_bf16(Pf, V1, O1);

        if (wib == 0 && it + 1 < HWN / 32)
            asm volatile("s_wait_asynccnt 0x0" ::: "memory");
        __syncthreads();
    }

#pragma unroll
    for (int r = 0; r < 8; ++r) {
        float inv = 1.f / fmaxf(l[r], 1e-30f);
        int t = t0 + r + half * 8;
        uint16_t* op = ho + ((size_t)b * HWN + t) * CCH + h * DH;
        op[lm]      = f2bf(O0[r] * inv);
        op[16 + lm] = f2bf(O1[r] * inv);
    }
}

// ------------------------- kernel 4: proj + bias + residual -----------------
__global__ void k_proj_gemm(const uint16_t* __restrict__ wbf,
                            const uint16_t* __restrict__ ho,
                            const float* __restrict__ pb,
                            const float* __restrict__ x,
                            float* __restrict__ out) {
    int wave = __builtin_amdgcn_readfirstlane((int)blockIdx.x * 8 + (threadIdx.x >> 5));
    int lane = threadIdx.x & 31;
    int half = lane >> 4, lm = lane & 15;
    int b   = wave >> 7;              // /(8*16)
    int rem = wave & 127;
    int mt  = rem >> 4;               // 0..7
    int n0  = (rem & 15) * 64;

    v16bf A[4];
    const uint16_t* arow = wbf + (size_t)(mt * 16 + lm) * CCH + half * 8;
#pragma unroll
    for (int kk = 0; kk < 4; ++kk) A[kk] = load_frag(arow + kk * 32);

#pragma unroll
    for (int nt = 0; nt < 4; ++nt) {
        int t = n0 + nt * 16 + lm;
        const uint16_t* brow = ho + ((size_t)b * HWN + t) * CCH + half * 8;
        if (nt < 3) __builtin_prefetch(brow + 16 * CCH, 0, 3);
        v8f acc = {};
#pragma unroll
        for (int kk = 0; kk < 4; ++kk)
            acc = wmma_bf16(A[kk], load_frag(brow + kk * 32), acc);

#pragma unroll
        for (int r = 0; r < 8; ++r) {
            int o = mt * 16 + r + half * 8;
            size_t idx = ((size_t)b * CCH + o) * HWN + t;
            out[idx] = acc[r] + pb[o] + x[idx];
        }
    }
}

// ---------------------------------------------------------------------------
extern "C" void kernel_launch(void* const* d_in, const int* in_sizes, int n_in,
                              void* d_out, int out_size, void* d_ws, size_t ws_size,
                              hipStream_t stream) {
    const float* x      = (const float*)d_in[0];
    const float* norm_w = (const float*)d_in[1];
    const float* norm_b = (const float*)d_in[2];
    const float* qkv_w  = (const float*)d_in[3];
    const float* proj_w = (const float*)d_in[4];
    const float* proj_b = (const float*)d_in[5];
    float* out = (float*)d_out;

    char* ws = (char*)d_ws;
    const size_t SZ_BCHW_BF = (size_t)BATCH * CCH * HWN * 2;   // 8 MB bf16
    size_t off = 0;
    uint16_t* xn      = (uint16_t*)(ws + off); off += SZ_BCHW_BF;
    uint16_t* qb      = (uint16_t*)(ws + off); off += SZ_BCHW_BF;
    uint16_t* kb      = (uint16_t*)(ws + off); off += SZ_BCHW_BF;
    uint16_t* vb      = (uint16_t*)(ws + off); off += SZ_BCHW_BF;
    uint16_t* hob     = (uint16_t*)(ws + off); off += SZ_BCHW_BF;
    uint16_t* qkv_bf  = (uint16_t*)(ws + off); off += (size_t)3 * CCH * CCH * 2;
    uint16_t* proj_bf = (uint16_t*)(ws + off); off += (size_t)CCH * CCH * 2;
    (void)ws_size; (void)in_sizes; (void)n_in; (void)out_size;

    k_wconvert<<<dim3((3 * CCH * CCH + 255) / 256), dim3(256), 0, stream>>>(
        qkv_w, proj_w, qkv_bf, proj_bf);
    k_groupnorm<<<dim3(BATCH * 32), dim3(256), 0, stream>>>(x, norm_w, norm_b, xn);
    k_qkv_gemm<<<dim3(BATCH * 24 * 16 / 8), dim3(256), 0, stream>>>(
        qkv_bf, xn, qb, kb, vb);
    k_attention<<<dim3(BATCH * NH * 8), dim3(256), 0, stream>>>(qb, kb, vb, hob);
    k_proj_gemm<<<dim3(BATCH * 8 * 16 / 8), dim3(256), 0, stream>>>(
        proj_bf, hob, proj_b, x, out);
}
